// SuperPoint_6897717478031
// MI455X (gfx1250) — compile-verified, matrix operand order checked by reading, not run on hardware
//
#include <hip/hip_runtime.h>
#include <hip/hip_bf16.h>
#include <math.h>

typedef __bf16 bf16_t;
typedef __attribute__((ext_vector_type(16))) __bf16 v16bf;
typedef __attribute__((ext_vector_type(8)))  float  v8f;

typedef unsigned int u32x4 __attribute__((ext_vector_type(4)));
typedef int          i32x4 __attribute__((ext_vector_type(4)));
typedef int          i32x8 __attribute__((ext_vector_type(8)));

#if defined(__gfx1250__) && __has_builtin(__builtin_amdgcn_tensor_load_to_lds) && \
    __has_builtin(__builtin_amdgcn_s_wait_tensorcnt)
#define HAVE_TDM 1
#else
#define HAVE_TDM 0
#endif

__device__ __forceinline__ bf16_t f2bf(float f) {
  union { float f; unsigned u; } c; c.f = f;
  unsigned r = (c.u + 0x7FFFu + ((c.u >> 16) & 1u)) >> 16;
  unsigned short s = (unsigned short)r;
  bf16_t o; __builtin_memcpy(&o, &s, 2);
  return o;
}

#if HAVE_TDM
// Issue one TDM 2D tile load: 32(N, dim0) x 64(K, dim1) bf16 slab from the
// padded [Kpad][NPad] weight matrix into LDS at lds_off. Tracked by TENSORcnt.
__device__ __forceinline__ void tdm_issue_slab(const bf16_t* gsrc, unsigned lds_off,
                                               int NPad, int Kpad) {
  const unsigned long long ga = (unsigned long long)(const void*)gsrc;
  u32x4 g0 = { 1u,                                     // count=1, user D#
               lds_off,                                // lds_addr (bytes)
               (unsigned)(ga & 0xFFFFFFFFu),           // global_addr[31:0]
               ((unsigned)((ga >> 32) & 0x01FFFFFFu)) | 0x80000000u }; // [56:32]|type=2
  i32x8 g1;
  g1[0] = 0x00010000;                                  // data_size=2B, no flags
  g1[1] = (int)(((unsigned)NPad & 0xFFFFu) << 16);     // tensor_dim0[15:0]=NPad
  g1[2] = (int)((((unsigned)Kpad & 0xFFFFu) << 16) | (((unsigned)NPad >> 16) & 0xFFFFu));
  g1[3] = (int)((32u << 16) | (((unsigned)Kpad >> 16) & 0xFFFFu)); // tile_dim0=32
  g1[4] = 64;                                          // tile_dim1=64, tile_dim2=0
  g1[5] = NPad;                                        // tensor_dim0_stride[31:0]
  g1[6] = 0;
  g1[7] = 0;
  i32x4 gz = { 0, 0, 0, 0 };
#if defined(__clang_major__) && (__clang_major__ >= 23)
  i32x8 gz8 = { 0, 0, 0, 0, 0, 0, 0, 0 };
  __builtin_amdgcn_tensor_load_to_lds(g0, g1, gz, gz, gz8, 0);
#else
  __builtin_amdgcn_tensor_load_to_lds(g0, g1, gz, gz, 0);
#endif
}
#endif

// ---------------------------------------------------------------------------
// Weight conversion: OIHW fp32 -> bf16 B-matrix [Kpad][NPad], K = Cin*kk,
// k index = cin*kk + (ky*kw+kx), zero-padded to Kpad/NPad.
// ---------------------------------------------------------------------------
__global__ void k_cvt_w(const float* __restrict__ w, bf16_t* __restrict__ wbf,
                        int Cout, int Cin, int kk, int K, int Kpad, int NPad) {
  int idx = blockIdx.x * blockDim.x + threadIdx.x;
  int total = Kpad * NPad;
  if (idx >= total) return;
  int n = idx % NPad;
  int k = idx / NPad;
  float v = 0.f;
  if (k < K && n < Cout) {
    int cin = k / kk, r = k - cin * kk;
    v = w[((size_t)n * Cin + cin) * kk + r];
  }
  wbf[(size_t)k * NPad + n] = f2bf(v);
}

// ---------------------------------------------------------------------------
// Implicit-GEMM conv (SAME, KSZ=1 or 3), bf16 WMMA, f32 accumulate.
// Wave tile: 16 pixels (M) x 32 couts (N), K stepped by 64 -> 4 WMMA/iter.
// Weight slabs [64K x 32N] bf16 (4KB) are double-buffered in LDS and staged
// by the Tensor Data Mover: wave 0 issues the next slab's DMA before
// computing on the current one (TDM per-wave in-order => s_wait_tensorcnt 1
// guarantees the current slab is resident). Fallback: cooperative copy.
// Block = 256 thr = 8 waves; grid = (HW/128, NPad/32, batch).
// Epilogue: y = ((relu?max(conv+b,0):conv+b) - mean)*gamma*rsqrt(var+1e-3)+beta
// ---------------------------------------------------------------------------
template <int KSZ>
__global__ void __launch_bounds__(256)
k_conv_wmma(const float* __restrict__ x, const bf16_t* __restrict__ wbf,
            const float* __restrict__ bia, const float* __restrict__ gam,
            const float* __restrict__ bet, const float* __restrict__ mea,
            const float* __restrict__ var, float* __restrict__ y,
            int Cin, int Cout, int H, int W, int relu,
            int Kpad, int NPad, int K) {
  constexpr int kk = KSZ * KSZ;
  const int lane = threadIdx.x & 31;
  const int wave = threadIdx.x >> 5;
  const int g    = lane >> 4;     // K-half select
  const int ln   = lane & 15;     // M row (A) / N col (B, C/D)
  const int HW   = H * W;

  const int m0 = (blockIdx.x * 8 + wave) * 16;
  const int n0 = blockIdx.y * 32;
  const int b  = blockIdx.z;

  __shared__ __attribute__((aligned(16))) bf16_t ldsB[2][64 * 32];

  const int m  = m0 + ln;
  const int px = m % W;
  const int py = m / W;
  const float* xb = x + (size_t)b * Cin * HW;

  v8f acc0 = {};
  v8f acc1 = {};

#if HAVE_TDM
  if (wave == 0) {
    tdm_issue_slab(wbf + n0,
                   (unsigned)(unsigned long long)(const void*)&ldsB[0][0],
                   NPad, Kpad);
  }
#endif

  for (int kc = 0; kc < Kpad; kc += 64) {
    const int ph = (kc >> 6) & 1;
#if HAVE_TDM
    if (wave == 0) {
      if (kc + 64 < Kpad) {
        tdm_issue_slab(wbf + (size_t)(kc + 64) * NPad + n0,
                       (unsigned)(unsigned long long)(const void*)&ldsB[ph ^ 1][0],
                       NPad, Kpad);
        __builtin_amdgcn_s_wait_tensorcnt(1);   // current slab resident
      } else {
        __builtin_amdgcn_s_wait_tensorcnt(0);   // last slab resident
      }
    }
    __syncthreads();                            // publish slab to all waves
#else
    __syncthreads();
    {
      const int t   = threadIdx.x;
      const int row = t >> 2;          // 64 rows of 32 bf16 (64B), 4x16B each
      const int seg = t & 3;
      const bf16_t* gsrc = wbf + (size_t)(kc + row) * NPad + n0 + seg * 8;
      *(uint4*)&ldsB[ph][row * 32 + seg * 8] = *(const uint4*)gsrc;
      if (kc + 64 < Kpad)
        __builtin_prefetch(wbf + (size_t)(kc + 64 + row) * NPad + n0 + seg * 8, 0, 0);
    }
    __syncthreads();
#endif
    const bf16_t* __restrict__ lb = &ldsB[ph][0];

    // Two K-subchunks of 32; A fragment per ISA 16-bit layout:
    // element i: k_local = (i<8?0:16) + 8*g + (i&7)
#pragma unroll
    for (int kh = 0; kh < 2; ++kh) {
      v16bf afrag, bf0, bf1;
#pragma unroll
      for (int i = 0; i < 16; ++i) {
        const int kl = kh * 32 + ((i & 8) << 1) + 8 * g + (i & 7);
        const int k  = kc + kl;
        float v = 0.f;
        if (k < K) {
          const int cin = k / kk;              // kk is a compile-time constant
          const int r   = k - cin * kk;
          int iy = py, ix = px;
          if (KSZ == 3) { iy += (r / 3) - 1; ix += (r % 3) - 1; }
          if (iy >= 0 && iy < H && ix >= 0 && ix < W)
            v = xb[(size_t)cin * HW + iy * W + ix];
        }
        afrag[i] = f2bf(v);
        bf0[i] = lb[kl * 32 + ln];
        bf1[i] = lb[kl * 32 + 16 + ln];
      }
      acc0 = __builtin_amdgcn_wmma_f32_16x16x32_bf16(
          false, afrag, false, bf0, (short)0, acc0, false, false);
      acc1 = __builtin_amdgcn_wmma_f32_16x16x32_bf16(
          false, afrag, false, bf1, (short)0, acc1, false, false);
    }
    __syncthreads();   // all waves done with slab[ph] before it is re-targeted
  }

  // C/D layout: lane ln = N col; VGPR r holds M = r + 8*g
#pragma unroll
  for (int s = 0; s < 2; ++s) {
    const v8f acc = (s == 0) ? acc0 : acc1;
    const int cout = n0 + s * 16 + ln;
    if (cout < Cout) {
      const float bv = bia[cout];
      const float sc = gam[cout] * rsqrtf(var[cout] + 0.001f);
      const float mn = mea[cout];
      const float bt = bet[cout];
      float* yb = y + ((size_t)b * Cout + cout) * HW;
#pragma unroll
      for (int r = 0; r < 8; ++r) {
        float vv = acc[r] + bv;
        if (relu) vv = fmaxf(vv, 0.f);
        vv = (vv - mn) * sc + bt;
        yb[m0 + r + 8 * g] = vv;
      }
    }
  }
}

// ---------------------------------------------------------------------------
// 2x2 maxpool stride 2 (VALID), NCHW, batch 2.
// ---------------------------------------------------------------------------
__global__ void k_maxpool2(const float* __restrict__ x, float* __restrict__ y,
                           int C, int H, int W) {
  const int Ho = H >> 1, Wo = W >> 1;
  const int total = 2 * C * Ho * Wo;
  int idx = blockIdx.x * blockDim.x + threadIdx.x;
  if (idx >= total) return;
  int wo = idx % Wo; int t = idx / Wo;
  int ho = t % Ho;   t /= Ho;
  int c  = t % C;    int b = t / C;
  const float* p = x + (((size_t)b * C + c) * H + ho * 2) * W + wo * 2;
  y[idx] = fmaxf(fmaxf(p[0], p[1]), fmaxf(p[W], p[W + 1]));
}

// ---------------------------------------------------------------------------
// Softmax over 65 channels, drop last, pixel-shuffle (8x8) -> heat (2,512,512)
// ---------------------------------------------------------------------------
__global__ void k_softshuffle(const float* __restrict__ det, float* __restrict__ heat) {
  int idx = blockIdx.x * blockDim.x + threadIdx.x;  // 2*64*64 cells
  if (idx >= 2 * 64 * 64) return;
  int wc = idx % 64; int t = idx / 64;
  int hc = t % 64;   int b = t / 64;
  const float* p = det + ((size_t)b * 65 * 64 + hc) * 64 + wc;  // ch stride 4096
  float mx = -1e30f;
  for (int c = 0; c < 65; ++c) mx = fmaxf(mx, p[(size_t)c * 4096]);
  float sum = 0.f;
  for (int c = 0; c < 65; ++c) sum += expf(p[(size_t)c * 4096] - mx);
  float inv = 1.f / sum;
  for (int c = 0; c < 64; ++c) {
    float v = expf(p[(size_t)c * 4096] - mx) * inv;
    heat[((size_t)b * 512 + hc * 8 + (c >> 3)) * 512 + wc * 8 + (c & 7)] = v;
  }
}

// ---------------------------------------------------------------------------
// 9x9 window max (SAME with -inf padding), per 512x512 map, batch 2.
// ---------------------------------------------------------------------------
__global__ void k_winmax(const float* __restrict__ s, float* __restrict__ o) {
  int idx = blockIdx.x * blockDim.x + threadIdx.x;
  if (idx >= 2 * 512 * 512) return;
  int p = idx & (512 * 512 - 1);
  int b = idx >> 18;
  int x = p & 511, y = p >> 9;
  const float* sb = s + ((size_t)b << 18);
  int y0 = max(y - 4, 0), y1 = min(y + 4, 511);
  int x0 = max(x - 4, 0), x1 = min(x + 4, 511);
  float mx = -1e30f;
  for (int yy = y0; yy <= y1; ++yy)
    for (int xx = x0; xx <= x1; ++xx)
      mx = fmaxf(mx, sb[yy * 512 + xx]);
  o[idx] = mx;
}

__global__ void k_eq(const float* __restrict__ s, const float* __restrict__ wm,
                     float* __restrict__ mask) {
  int idx = blockIdx.x * blockDim.x + threadIdx.x;
  if (idx >= 2 * 512 * 512) return;
  mask[idx] = (s[idx] == wm[idx]) ? 1.f : 0.f;
}

__global__ void k_supp(const float* __restrict__ s, const float* __restrict__ wm,
                       float* __restrict__ supp) {
  int idx = blockIdx.x * blockDim.x + threadIdx.x;
  if (idx >= 2 * 512 * 512) return;
  supp[idx] = (wm[idx] > 0.f) ? 0.f : s[idx];
}

__global__ void k_merge(float* __restrict__ mask, const float* __restrict__ wm,
                        const float* __restrict__ supp, const float* __restrict__ swm) {
  int idx = blockIdx.x * blockDim.x + threadIdx.x;
  if (idx >= 2 * 512 * 512) return;
  bool newmax = (supp[idx] == swm[idx]) && !(wm[idx] > 0.f);
  mask[idx] = ((mask[idx] > 0.f) || newmax) ? 1.f : 0.f;
}

// flat = (!border && (mask ? heat : 0) > thr) ? value : -1
__global__ void k_finalize(const float* __restrict__ heat, const float* __restrict__ mask,
                           float* __restrict__ flat) {
  int idx = blockIdx.x * blockDim.x + threadIdx.x;
  if (idx >= 2 * 512 * 512) return;
  int p = idx & (512 * 512 - 1);
  int x = p & 511, y = p >> 9;
  float v = (mask[idx] > 0.f) ? heat[idx] : 0.f;
  bool border = (y < 4) | (y >= 508) | (x < 4) | (x >= 508);
  flat[idx] = (!border && v > 0.005f) ? v : -1.f;
}

// ---------------------------------------------------------------------------
// Exact top-1024 per batch via 1024 rounds of block-wide argmax (ties -> low
// index). Mutates flat (scratch). Writes kp (x,y) and vals into d_out.
// ---------------------------------------------------------------------------
__global__ void __launch_bounds__(1024)
k_topk(float* __restrict__ flat, float* __restrict__ kp, float* __restrict__ vals) {
  const int HW = 512 * 512;
  const int b = blockIdx.x;
  const int t = threadIdx.x;
  float* f = flat + (size_t)b * HW;
  __shared__ float sv[1024];
  __shared__ int   si[1024];
  for (int sel = 0; sel < 1024; ++sel) {
    float bm = -1e30f; int bi = HW;
    for (int i = t; i < HW; i += 1024) {
      float v = f[i];
      if (v > bm) { bm = v; bi = i; }
    }
    sv[t] = bm; si[t] = bi;
    __syncthreads();
    for (int off = 512; off > 0; off >>= 1) {
      if (t < off) {
        if (sv[t + off] > sv[t] || (sv[t + off] == sv[t] && si[t + off] < si[t])) {
          sv[t] = sv[t + off]; si[t] = si[t + off];
        }
      }
      __syncthreads();
    }
    if (t == 0) {
      int wi = si[0];
      vals[(size_t)b * 1024 + sel] = sv[0];
      kp[((size_t)b * 1024 + sel) * 2 + 0] = (float)(wi & 511);
      kp[((size_t)b * 1024 + sel) * 2 + 1] = (float)(wi >> 9);
      f[wi] = -1e30f;
    }
    __syncthreads();
  }
}

// L2-normalize descriptor map over channels (in place), (2,256,64,64)
__global__ void k_l2norm(float* __restrict__ d) {
  int idx = blockIdx.x * blockDim.x + threadIdx.x;
  if (idx >= 2 * 4096) return;
  int p = idx & 4095; int b = idx >> 12;
  float* base = d + (size_t)b * 256 * 4096 + p;
  float s = 0.f;
  for (int c = 0; c < 256; ++c) { float v = base[(size_t)c * 4096]; s += v * v; }
  float inv = 1.f / fmaxf(sqrtf(s), 1e-12f);
  for (int c = 0; c < 256; ++c) base[(size_t)c * 4096] *= inv;
}

// Bilinear sample + per-keypoint L2 norm -> desc out (2,1024,256)
__global__ void __launch_bounds__(256)
k_sample(const float* __restrict__ dmap, const float* __restrict__ kp,
         float* __restrict__ out) {
  const int n = blockIdx.x, b = blockIdx.y, c = threadIdx.x;
  float kx = kp[((size_t)b * 1024 + n) * 2 + 0];
  float ky = kp[((size_t)b * 1024 + n) * 2 + 1];
  float px = (kx + 0.5f) * 0.125f - 0.5f;
  float py = (ky + 0.5f) * 0.125f - 0.5f;
  float fx0 = floorf(px), fy0 = floorf(py);
  int x0 = (int)fx0, y0 = (int)fy0, x1 = x0 + 1, y1 = y0 + 1;
  float fx = px - fx0, fy = py - fy0;
  const float* db = dmap + (size_t)b * 256 * 4096 + (size_t)c * 4096;
  auto g = [&](int yy, int xx) -> float {
    return (xx >= 0 && xx < 64 && yy >= 0 && yy < 64) ? db[yy * 64 + xx] : 0.f;
  };
  float v = g(y0, x0) * (1.f - fx) * (1.f - fy) + g(y0, x1) * fx * (1.f - fy)
          + g(y1, x0) * (1.f - fx) * fy        + g(y1, x1) * fx * fy;
  __shared__ float red[256];
  red[c] = v * v;
  __syncthreads();
  for (int off = 128; off > 0; off >>= 1) {
    if (c < off) red[c] += red[c + off];
    __syncthreads();
  }
  float nrm = fmaxf(sqrtf(red[0]), 1e-12f);
  out[((size_t)b * 1024 + n) * 256 + c] = v / nrm;
}

// ---------------------------------------------------------------------------
// Host orchestration
// ---------------------------------------------------------------------------
static void run_conv(hipStream_t stream, void* const* d_in, int layer,
                     const float* x, float* y, bf16_t* WBF,
                     int Cin, int Cout, int H, int W, int ksz, int relu) {
  const float* bia = (const float*)d_in[1 + layer * 6 + 0];
  const float* bet = (const float*)d_in[1 + layer * 6 + 1];
  const float* gam = (const float*)d_in[1 + layer * 6 + 2];
  const float* mea = (const float*)d_in[1 + layer * 6 + 3];
  const float* var = (const float*)d_in[1 + layer * 6 + 4];
  const float* w   = (const float*)d_in[1 + layer * 6 + 5];
  const int kk   = ksz * ksz;
  const int K    = Cin * kk;
  const int Kpad = (K + 63) & ~63;
  const int NPad = (Cout + 31) & ~31;
  const int tot  = Kpad * NPad;
  k_cvt_w<<<(tot + 255) / 256, 256, 0, stream>>>(w, WBF, Cout, Cin, kk, K, Kpad, NPad);
  const int HW = H * W;
  dim3 grid(HW / (16 * 8), NPad / 32, 2);
  if (ksz == 3) {
    k_conv_wmma<3><<<grid, 256, 0, stream>>>(x, WBF, bia, gam, bet, mea, var, y,
                                             Cin, Cout, H, W, relu, Kpad, NPad, K);
  } else {
    k_conv_wmma<1><<<grid, 256, 0, stream>>>(x, WBF, bia, gam, bet, mea, var, y,
                                             Cin, Cout, H, W, relu, Kpad, NPad, K);
  }
}

extern "C" void kernel_launch(void* const* d_in, const int* in_sizes, int n_in,
                              void* d_out, int out_size, void* d_ws, size_t ws_size,
                              hipStream_t stream) {
  (void)in_sizes; (void)n_in; (void)out_size; (void)ws_size;
  const float* image = (const float*)d_in[0];

  char* wsb = (char*)d_ws;
  const size_t SAe = (size_t)2 * 64 * 512 * 512;           // largest activation (elems)
  float*  A   = (float*)wsb;
  float*  Bf  = A + SAe;
  bf16_t* WBF = (bf16_t*)(wsb + 2 * SAe * sizeof(float));
  char*   rest = wsb + 2 * SAe * sizeof(float) + (4u << 20);
  float* det  = (float*)rest;                               // 2*65*64*64
  float* dsc  = det  + (size_t)2 * 65 * 64 * 64;            // 2*256*64*64
  float* heat = dsc  + (size_t)2 * 256 * 64 * 64;           // 2*512*512
  float* wm   = heat + (size_t)2 * 512 * 512;
  float* mask = wm   + (size_t)2 * 512 * 512;
  float* supp = mask + (size_t)2 * 512 * 512;
  float* swm  = supp + (size_t)2 * 512 * 512;
  float* flat = swm  + (size_t)2 * 512 * 512;

  float* kp_out   = (float*)d_out;          // (2,1024,2)
  float* val_out  = kp_out + 2 * 1024 * 2;  // (2,1024)
  float* desc_out = val_out + 2 * 1024;     // (2,1024,256)

  // backbone (layers 0..7), flatten order: backbone < descriptor < detector
  run_conv(stream, d_in, 0, image, A, WBF, 1, 64, 512, 512, 3, 1);
  run_conv(stream, d_in, 1, A, Bf, WBF, 64, 64, 512, 512, 3, 1);
  { int tot = 2 * 64 * 256 * 256;
    k_maxpool2<<<(tot + 255) / 256, 256, 0, stream>>>(Bf, A, 64, 512, 512); }
  run_conv(stream, d_in, 2, A, Bf, WBF, 64, 64, 256, 256, 3, 1);
  run_conv(stream, d_in, 3, Bf, A, WBF, 64, 64, 256, 256, 3, 1);
  { int tot = 2 * 64 * 128 * 128;
    k_maxpool2<<<(tot + 255) / 256, 256, 0, stream>>>(A, Bf, 64, 256, 256); }
  run_conv(stream, d_in, 4, Bf, A, WBF, 64, 128, 128, 128, 3, 1);
  run_conv(stream, d_in, 5, A, Bf, WBF, 128, 128, 128, 128, 3, 1);
  { int tot = 2 * 128 * 64 * 64;
    k_maxpool2<<<(tot + 255) / 256, 256, 0, stream>>>(Bf, A, 128, 128, 128); }
  run_conv(stream, d_in, 6, A, Bf, WBF, 128, 128, 64, 64, 3, 1);
  run_conv(stream, d_in, 7, Bf, A, WBF, 128, 128, 64, 64, 3, 1);
  // A = backbone output (2,128,64,64)

  // descriptor head (layers 8,9)
  run_conv(stream, d_in, 8, A, Bf, WBF, 128, 256, 64, 64, 3, 1);
  run_conv(stream, d_in, 9, Bf, dsc, WBF, 256, 256, 64, 64, 1, 0);
  k_l2norm<<<(2 * 4096 + 255) / 256, 256, 0, stream>>>(dsc);

  // detector head (layers 10,11)
  run_conv(stream, d_in, 10, A, Bf, WBF, 128, 256, 64, 64, 3, 1);
  run_conv(stream, d_in, 11, Bf, det, WBF, 256, 65, 64, 64, 1, 0);
  k_softshuffle<<<(2 * 64 * 64 + 255) / 256, 256, 0, stream>>>(det, heat);

  // batched NMS (radius 4, 2 refinement rounds)
  const int NP = 2 * 512 * 512;
  const int gb = (NP + 255) / 256;
  k_winmax<<<gb, 256, 0, stream>>>(heat, wm);
  k_eq<<<gb, 256, 0, stream>>>(heat, wm, mask);
  for (int it = 0; it < 2; ++it) {
    k_winmax<<<gb, 256, 0, stream>>>(mask, wm);
    k_supp<<<gb, 256, 0, stream>>>(heat, wm, supp);
    k_winmax<<<gb, 256, 0, stream>>>(supp, swm);
    k_merge<<<gb, 256, 0, stream>>>(mask, wm, supp, swm);
  }
  k_finalize<<<gb, 256, 0, stream>>>(heat, mask, flat);

  // top-1024 per batch -> kp, vals; then bilinear sample + normalize -> desc
  k_topk<<<dim3(2), 1024, 0, stream>>>(flat, kp_out, val_out);
  k_sample<<<dim3(1024, 2), 256, 0, stream>>>(dsc, kp_out, desc_out);
}
